// FCOSHead_60249801228382
// MI455X (gfx1250) — matrix operand order repaired and checked by reading, not compiled, not run on hardware
//
#include <hip/hip_runtime.h>

typedef __bf16 bf16;
typedef __attribute__((ext_vector_type(16))) __bf16 v16bf;
typedef __attribute__((ext_vector_type(8)))  __bf16 v8bf;
typedef __attribute__((ext_vector_type(8)))  float  v8f;

#define WMMA_BF16(a, b, c) \
  __builtin_amdgcn_wmma_f32_16x16x32_bf16(false, (a), false, (b), (short)0, (c), false, false)

// CDNA5 async global->LDS copy (ASYNCcnt-tracked DMA), 16B per lane.
__device__ __forceinline__ void async_ld_b128(unsigned lds_off, const void* g) {
  asm volatile("global_load_async_to_lds_b128 %0, %1, off"
               :: "v"(lds_off), "v"(g) : "memory");
}
__device__ __forceinline__ void s_wait_async0() {
  asm volatile("s_wait_asynccnt 0" ::: "memory");
}

// ---------------------------------------------------------------------------
// Prep: NCHW fp32 -> NHWC bf16
// ---------------------------------------------------------------------------
__global__ __launch_bounds__(256) void nchw_to_nhwc_bf16(
    const float* __restrict__ in, bf16* __restrict__ out, int H, int W) {
  size_t idx = (size_t)blockIdx.x * 256 + threadIdx.x;
  size_t HW = (size_t)H * W;
  size_t total = (size_t)8 * 256 * HW;
  if (idx >= total) return;
  int c = (int)(idx & 255);
  size_t r = idx >> 8;            // n*HW + s
  int n = (int)(r / HW);
  size_t s = r % HW;
  out[idx] = (bf16)in[((size_t)n * 256 + c) * HW + s];
}

// ---------------------------------------------------------------------------
// Prep: tower weights -> B-fragment layout [br][stage][kc(72)][nt(16)][lane(32)][e(16)]
//   element = B[K = kc*32 + (e/8)*16 + (lane/16)*8 + e%8][N = nt*16 + lane%16]
//   B[K][N] = w[N][ci][tap],  K = tap*256 + ci
// ---------------------------------------------------------------------------
__global__ __launch_bounds__(256) void prep_tower_w(
    const float* __restrict__ cls_w, const float* __restrict__ reg_w,
    bf16* __restrict__ frag) {
  size_t idx = (size_t)blockIdx.x * 256 + threadIdx.x;
  const size_t total = (size_t)2 * 4 * 72 * 16 * 32 * 16;
  if (idx >= total) return;
  int e    = (int)(idx & 15);
  int lane = (int)((idx >> 4) & 31);
  int nt   = (int)((idx >> 9) & 15);
  size_t r = idx >> 13;
  int kc   = (int)(r % 72);
  size_t r2 = r / 72;
  int stage = (int)(r2 & 3);
  int br    = (int)(r2 >> 2);
  int K   = kc * 32 + ((e >> 3) << 4) + ((lane >> 4) << 3) + (e & 7);
  int N   = nt * 16 + (lane & 15);
  int tap = K >> 8;
  int ci  = K & 255;
  const float* src = (br == 0) ? cls_w : reg_w;
  frag[idx] = (bf16)src[(((size_t)stage * 256 + N) * 256 + ci) * 9 + tap];
}

// fcls_w [80][256][3][3] -> [kc(72)][nt(5)][lane(32)][e(16)]
__global__ __launch_bounds__(256) void prep_cls_w(
    const float* __restrict__ fcls_w, bf16* __restrict__ frag) {
  size_t idx = (size_t)blockIdx.x * 256 + threadIdx.x;
  const size_t total = (size_t)72 * 5 * 32 * 16;
  if (idx >= total) return;
  int e    = (int)(idx & 15);
  int lane = (int)((idx >> 4) & 31);
  size_t r = idx >> 9;
  int nt   = (int)(r % 5);
  int kc   = (int)(r / 5);
  int K   = kc * 32 + ((e >> 3) << 4) + ((lane >> 4) << 3) + (e & 7);
  int N   = nt * 16 + (lane & 15);           // < 80 always
  int tap = K >> 8;
  int ci  = K & 255;
  frag[idx] = (bf16)fcls_w[((size_t)N * 256 + ci) * 9 + tap];
}

// merged reg(4)+ctr(1) final weights -> [kc(72)][lane(32)][e(16)], cols 5..15 = 0
__global__ __launch_bounds__(256) void prep_regctr_w(
    const float* __restrict__ freg_w, const float* __restrict__ fctr_w,
    bf16* __restrict__ frag) {
  size_t idx = (size_t)blockIdx.x * 256 + threadIdx.x;
  const size_t total = (size_t)72 * 32 * 16;
  if (idx >= total) return;
  int e    = (int)(idx & 15);
  int lane = (int)((idx >> 4) & 31);
  int kc   = (int)(idx >> 9);
  int K   = kc * 32 + ((e >> 3) << 4) + ((lane >> 4) << 3) + (e & 7);
  int N   = lane & 15;
  int tap = K >> 8;
  int ci  = K & 255;
  float v = 0.f;
  if (N < 4)       v = freg_w[((size_t)N * 256 + ci) * 9 + tap];
  else if (N == 4) v = fctr_w[(size_t)ci * 9 + tap];
  frag[idx] = (bf16)v;
}

// ---------------------------------------------------------------------------
// Tower conv: 3x3, 256->256, bias + ReLU, NHWC bf16 in/out.
// WG = 256 thr = 8 waves (2 M-wave-groups x 4 N-wave-groups).
// WG tile: M=128 spatial, N=256. Per-wave register tile: 64x64
// (4 M-subtiles x 4 N-tiles) so each B fragment feeds 4 WMMAs.
// Per-tap 128x256 A slab, double-buffered, staged by async global->LDS DMA.
// A row stride padded to 264 bf16 for conflict-minimal ds_load_b128.
// ---------------------------------------------------------------------------
#define APAD 264

__global__ __launch_bounds__(256) void conv3x3_tower(
    const bf16* __restrict__ X, const bf16* __restrict__ Wf,
    const float* __restrict__ bias, bf16* __restrict__ Y, int H, int W) {
  __shared__ union SM {
    bf16 A[2][128][APAD];   // 2 x 66KB tap slabs
    bf16 O[128][256];       // epilogue staging (overlays A[0]; used after last read)
  } sm;

  const int tid    = threadIdx.x;
  const int lane   = tid & 31;
  const int wave   = tid >> 5;
  const int wave_m = wave & 1;    // 0..1  -> M offset 0/64
  const int wave_n = wave >> 1;   // 0..3  -> N offset 0/64/128/192
  const int HW     = H * W;
  const int n_img  = blockIdx.y;
  const int m0     = blockIdx.x * 128;

  // staging role: row sr (0..127), 256B chunk sp (0..1)
  const int sr = tid >> 1;
  const int sp = tid & 1;
  const int s_pos = m0 + sr;
  const int sh = s_pos / W;
  const int sw = s_pos - sh * W;
  const bool srow_ok = s_pos < HW;

  const int nl = lane & 15;
  const int g  = lane >> 4;
  const int mrow = wave_m * 64 + nl;   // M-subtiles at mrow + {0,16,32,48}

  v8f acc[4][4] = {};

  // stage tap slab into buffer `buf` via async DMA (zeros via ds_store if OOB)
  auto stage = [&](int tap, int buf) {
    const int hh = sh + tap / 3 - 1;
    const int ww = sw + tap % 3 - 1;
    const bool v = srow_ok && hh >= 0 && hh < H && ww >= 0 && ww < W;
    bf16* dstp = &sm.A[buf][sr][sp * 128];
    if (v) {
      const bf16* src = X + (((size_t)n_img * H + hh) * W + ww) * 256 + sp * 128;
      const unsigned dlo = (unsigned)(uintptr_t)dstp;
#pragma unroll
      for (int q = 0; q < 16; ++q)
        async_ld_b128(dlo + q * 16, src + q * 8);
    } else {
      const uint4 z = make_uint4(0u, 0u, 0u, 0u);
#pragma unroll
      for (int q = 0; q < 16; ++q)
        *(uint4*)(dstp + q * 8) = z;
    }
  };

  stage(0, 0);
  s_wait_async0();
  __syncthreads();

  for (int tap = 0; tap < 9; ++tap) {
    const int cur = tap & 1;
    if (tap < 8) stage(tap + 1, cur ^ 1);   // DMA next tap while computing this one

    const bf16* arow = &sm.A[cur][mrow][0];
    for (int cc = 0; cc < 8; ++cc) {
      const int co = cc * 32;
      v16bf a[4];
#pragma unroll
      for (int mi = 0; mi < 4; ++mi) {
        const bf16* ap = arow + (size_t)mi * 16 * APAD + co;
        v8bf lo = *(const v8bf*)(ap + g * 8);
        v8bf hi = *(const v8bf*)(ap + 16 + g * 8);
        a[mi] = __builtin_shufflevector(lo, hi,
            0, 1, 2, 3, 4, 5, 6, 7, 8, 9, 10, 11, 12, 13, 14, 15);
      }
      const int kc = tap * 8 + cc;
      const bf16* wb = Wf + ((size_t)kc * 16 * 32) * 16;
#pragma unroll
      for (int t = 0; t < 4; ++t) {
        const int nt = wave_n * 4 + t;
        v16bf b = *(const v16bf*)(wb + ((size_t)nt * 32 + lane) * 16);
#pragma unroll
        for (int mi = 0; mi < 4; ++mi)
          acc[mi][t] = WMMA_BF16(a[mi], b, acc[mi][t]);
      }
    }
    s_wait_async0();       // our async writes to the other buffer are visible
    __syncthreads();       // everyone done reading cur / writing next
  }

  // epilogue: bias + ReLU -> LDS (overlays A) -> coalesced bf16 stores
#pragma unroll
  for (int mi = 0; mi < 4; ++mi) {
#pragma unroll
    for (int t = 0; t < 4; ++t) {
      const int col = wave_n * 64 + t * 16 + nl;
      const float bs = bias[col];
#pragma unroll
      for (int vv = 0; vv < 8; ++vv) {
        const int row = wave_m * 64 + mi * 16 + vv + 8 * g;
        float val = acc[mi][t][vv] + bs;
        val = val > 0.f ? val : 0.f;
        sm.O[row][col] = (bf16)val;
      }
    }
  }
  __syncthreads();
  if (srow_ok) {
    bf16* dst = Y + (((size_t)n_img * H + sh) * W + sw) * 256 + sp * 128;
    const bf16* so = &sm.O[sr][sp * 128];
#pragma unroll
    for (int q = 0; q < 16; ++q)
      *(uint4*)(dst + q * 8) = *(const uint4*)(so + q * 8);
  }
}

// ---------------------------------------------------------------------------
// Final conv: 3x3, 256 -> NT*16 cols, writes fp32 NCHW into d_out.
// WG = 8 M-waves, tile M=128. MODE 0: cls (+bias). MODE 1: reg exp / ctr.
// ---------------------------------------------------------------------------
template <int NT, int MODE>
__global__ __launch_bounds__(256) void conv3x3_final(
    const bf16* __restrict__ X, const bf16* __restrict__ Wf,
    const float* __restrict__ b0, const float* __restrict__ b1,
    const float* __restrict__ reg_coef, int level,
    float* __restrict__ out, int H, int W,
    size_t off0, size_t off1, int NC0) {
  __shared__ bf16 Alds[128][32];
  const int tid  = threadIdx.x;
  const int lane = tid & 31;
  const int wave = tid >> 5;
  const int HW   = H * W;
  const int n_img = blockIdx.y;
  const int m0   = blockIdx.x * 128;

  const int sr = tid >> 1;   // 0..127
  const int sp = tid & 1;    // 16 bf16 each
  const int s_pos = m0 + sr;
  const int sh = s_pos / W;
  const int sw = s_pos - sh * W;
  const bool srow_ok = s_pos < HW;

  const int nl = lane & 15;
  const int g  = lane >> 4;
  const int mbase = wave * 16;

  v8f acc[NT] = {};

  for (int tap = 0; tap < 9; ++tap) {
    const int hh = sh + tap / 3 - 1;
    const int ww = sw + tap % 3 - 1;
    const bool v = srow_ok && hh >= 0 && hh < H && ww >= 0 && ww < W;
    const bf16* src = X + (((size_t)n_img * H + hh) * W + ww) * 256;
    for (int cc = 0; cc < 8; ++cc) {
      __syncthreads();
      uint4 d0 = make_uint4(0u, 0u, 0u, 0u), d1 = make_uint4(0u, 0u, 0u, 0u);
      if (v) {
        d0 = *(const uint4*)(src + cc * 32 + sp * 16);
        d1 = *(const uint4*)(src + cc * 32 + sp * 16 + 8);
      }
      *(uint4*)(&Alds[sr][sp * 16])     = d0;
      *(uint4*)(&Alds[sr][sp * 16 + 8]) = d1;
      __syncthreads();
      v8bf alo = *(const v8bf*)(&Alds[mbase + nl][g * 8]);
      v8bf ahi = *(const v8bf*)(&Alds[mbase + nl][16 + g * 8]);
      v16bf a = __builtin_shufflevector(alo, ahi,
          0, 1, 2, 3, 4, 5, 6, 7, 8, 9, 10, 11, 12, 13, 14, 15);
      const int kc = tap * 8 + cc;
#pragma unroll
      for (int t = 0; t < NT; ++t) {
        v16bf b = *(const v16bf*)(Wf + (((size_t)kc * NT + t) * 32 + lane) * 16);
        acc[t] = WMMA_BF16(a, b, acc[t]);
      }
    }
  }

  const float coef = (MODE == 1) ? reg_coef[level] : 0.f;
#pragma unroll
  for (int t = 0; t < NT; ++t) {
#pragma unroll
    for (int vv = 0; vv < 8; ++vv) {
      const int m = mbase + vv + 8 * g;
      const int s = m0 + m;
      if (s < HW) {
        const int h = s / W;
        const int w = s - h * W;
        const int co = t * 16 + nl;
        const float val = acc[t][vv];
        if (MODE == 0) {
          out[off0 + (((size_t)n_img * NC0 + co) * H + h) * W + w] = val + b0[co];
        } else {
          if (co < 4)
            out[off0 + (((size_t)n_img * 4 + co) * H + h) * W + w] =
                __expf((val + b0[co]) * coef);
          else if (co == 4)
            out[off1 + ((size_t)n_img * H + h) * W + w] = val + b1[0];
        }
      }
    }
  }
}

// ---------------------------------------------------------------------------
// Host launcher
// ---------------------------------------------------------------------------
extern "C" void kernel_launch(void* const* d_in, const int* in_sizes, int n_in,
                              void* d_out, int out_size, void* d_ws, size_t ws_size,
                              hipStream_t stream) {
  const float* x[5] = {(const float*)d_in[0], (const float*)d_in[1],
                       (const float*)d_in[2], (const float*)d_in[3],
                       (const float*)d_in[4]};
  const float* cls_w  = (const float*)d_in[5];
  const float* cls_b  = (const float*)d_in[6];
  const float* reg_w  = (const float*)d_in[7];
  const float* reg_b  = (const float*)d_in[8];
  const float* fcls_w = (const float*)d_in[9];
  const float* fcls_b = (const float*)d_in[10];
  const float* freg_w = (const float*)d_in[11];
  const float* freg_b = (const float*)d_in[12];
  const float* fctr_w = (const float*)d_in[13];
  const float* fctr_b = (const float*)d_in[14];
  const float* rcoef  = (const float*)d_in[15];
  float* out = (float*)d_out;

  const int Hs[5] = {100, 50, 25, 13, 7};
  const int Wsz[5] = {152, 76, 38, 19, 10};

  size_t off = 0;
  auto carve = [&](size_t bytes) {
    size_t o = off;
    off = (off + bytes + 255) & ~(size_t)255;
    return o;
  };
  char* ws = (char*)d_ws;
  const size_t towerFragElems  = (size_t)2 * 4 * 72 * 16 * 32 * 16;
  const size_t clsFragElems    = (size_t)72 * 5 * 32 * 16;
  const size_t rcFragElems     = (size_t)72 * 32 * 16;
  const size_t actBytes        = (size_t)8 * 100 * 152 * 256 * sizeof(bf16);
  bf16* fragTower = (bf16*)(ws + carve(towerFragElems * 2));
  bf16* fragCls   = (bf16*)(ws + carve(clsFragElems * 2));
  bf16* fragRC    = (bf16*)(ws + carve(rcFragElems * 2));
  bf16* Xb = (bf16*)(ws + carve(actBytes));
  bf16* T0 = (bf16*)(ws + carve(actBytes));
  bf16* T1 = (bf16*)(ws + carve(actBytes));

  prep_tower_w<<<(int)((towerFragElems + 255) / 256), 256, 0, stream>>>(cls_w, reg_w, fragTower);
  prep_cls_w<<<(int)((clsFragElems + 255) / 256), 256, 0, stream>>>(fcls_w, fragCls);
  prep_regctr_w<<<(int)((rcFragElems + 255) / 256), 256, 0, stream>>>(freg_w, fctr_w, fragRC);

  // output tuple offsets: 5 cls, then 5 reg, then 5 ctr
  size_t cls_off[5], reg_off[5], ctr_off[5];
  size_t tot_cls = 0, tot_reg = 0;
  for (int l = 0; l < 5; ++l) {
    size_t hw = (size_t)Hs[l] * Wsz[l];
    tot_cls += 8 * 80 * hw;
    tot_reg += 8 * 4 * hw;
  }
  {
    size_t a = 0, b = tot_cls, c = tot_cls + tot_reg;
    for (int l = 0; l < 5; ++l) {
      size_t hw = (size_t)Hs[l] * Wsz[l];
      cls_off[l] = a; a += 8 * 80 * hw;
      reg_off[l] = b; b += 8 * 4 * hw;
      ctr_off[l] = c; c += 8 * 1 * hw;
    }
  }

  const size_t stageFrag = (size_t)72 * 16 * 32 * 16;  // elems per (branch,stage)

  for (int l = 0; l < 5; ++l) {
    const int H = Hs[l], W = Wsz[l];
    const int HW = H * W;
    size_t total = (size_t)8 * 256 * HW;
    nchw_to_nhwc_bf16<<<(int)((total + 255) / 256), 256, 0, stream>>>(x[l], Xb, H, W);

    dim3 gT((HW + 127) / 128, 8);
    dim3 gF((HW + 127) / 128, 8);

    // cls tower chain: X -> T0 -> T1 -> T0 -> T1
    {
      const bf16* cur = Xb;
      for (int s = 0; s < 4; ++s) {
        bf16* nxt = (s & 1) ? T1 : T0;
        conv3x3_tower<<<gT, 256, 0, stream>>>(cur, fragTower + (0 * 4 + s) * stageFrag,
                                              cls_b + s * 256, nxt, H, W);
        cur = nxt;
      }
      conv3x3_final<5, 0><<<gF, 256, 0, stream>>>(cur, fragCls, fcls_b, nullptr,
                                                  rcoef, l, out, H, W,
                                                  cls_off[l], (size_t)0, 80);
    }
    // reg tower chain (reuses T0/T1)
    {
      const bf16* cur = Xb;
      for (int s = 0; s < 4; ++s) {
        bf16* nxt = (s & 1) ? T1 : T0;
        conv3x3_tower<<<gT, 256, 0, stream>>>(cur, fragTower + (1 * 4 + s) * stageFrag,
                                              reg_b + s * 256, nxt, H, W);
        cur = nxt;
      }
      conv3x3_final<1, 1><<<gF, 256, 0, stream>>>(cur, fragRC, freg_b, fctr_b,
                                                  rcoef, l, out, H, W,
                                                  reg_off[l], ctr_off[l], 4);
    }
  }
}
